// MultiHeadAttention_42949673691
// MI455X (gfx1250) — matrix operand order repaired and checked by reading, not compiled, not run on hardware
//
#include <hip/hip_runtime.h>

#define D_MODEL 1024
#define NHEAD 16
#define DK 64
#define BATCH 4
#define SEQLEN 2048

typedef __attribute__((ext_vector_type(16))) __bf16        v16bf;
typedef __attribute__((ext_vector_type(2)))  __bf16        v2bf;
typedef __attribute__((ext_vector_type(8)))  float         v8f;
typedef __attribute__((ext_vector_type(8)))  unsigned int  v8u;
typedef __attribute__((ext_vector_type(4)))  unsigned int  u32x4;
typedef __attribute__((ext_vector_type(8)))  int           i32x8;
typedef __attribute__((ext_vector_type(4)))  int           i32x4;

#if defined(__HIP_DEVICE_COMPILE__) && __has_builtin(__builtin_amdgcn_tensor_load_to_lds)
#define USE_TDM 1
#else
#define USE_TDM 0
#endif

__device__ __forceinline__ unsigned short f2bf(float f) {
  unsigned int u = __builtin_bit_cast(unsigned int, f);
  u += 0x7FFFu + ((u >> 16) & 1u);            // round-to-nearest-even
  return (unsigned short)(u >> 16);
}

// packed pair convert (v_cvt_pk_bf16_f32 when available)
__device__ __forceinline__ unsigned int f2bf2(float lo, float hi) {
#if defined(__HIP_DEVICE_COMPILE__) && __has_builtin(__builtin_amdgcn_cvt_pk_bf16_f32)
  v2bf p = __builtin_amdgcn_cvt_pk_bf16_f32(lo, hi);
  return __builtin_bit_cast(unsigned int, p);
#else
  return (unsigned int)f2bf(lo) | ((unsigned int)f2bf(hi) << 16);
#endif
}

// Build a 16xbf16 fragment from two 16-byte LDS loads (ds_load_b128 x2)
__device__ __forceinline__ v16bf ld_frag(const unsigned short* p0, const unsigned short* p1) {
  uint4 x = *(const uint4*)p0;
  uint4 y = *(const uint4*)p1;
  v8u u = {x.x, x.y, x.z, x.w, y.x, y.y, y.z, y.w};
  return __builtin_bit_cast(v16bf, u);
}

#define WMMA_BF16(A, B, C) \
  __builtin_amdgcn_wmma_f32_16x16x32_bf16(false, (A), false, (B), (short)0, (C), false, false)

#if USE_TDM
// TDM 2D bf16 tile load: rows x tile_k elems, memory row stride stride_elems,
// LDS rows padded by (pad_amt+1)*4 bytes after each 2^(pad_int+3) bytes.
// Does NOT wait; caller must s_wait_tensorcnt + barrier before consumption.
__device__ __forceinline__ void tdm_load_2d(const unsigned short* gsrc,
                                            unsigned lds_byte_off,
                                            unsigned tile_k, unsigned rows,
                                            unsigned stride_elems,
                                            unsigned pad_int, unsigned pad_amt) {
  unsigned long long ga = (unsigned long long)(uintptr_t)gsrc;
  u32x4 g0;
  g0[0] = 1u;                                         // count=1, user mode
  g0[1] = lds_byte_off;                               // D#.lds_addr
  g0[2] = (unsigned)(ga & 0xFFFFFFFFu);               // global_addr[31:0]
  g0[3] = (unsigned)((ga >> 32) & 0x01FFFFFFu)        // global_addr[56:32]
          | (2u << 30);                               // type=2 ("image")
  i32x8 g1;
  g1[0] = (int)((1u << 16)                            // data_size = 2 bytes
              | (1u << 20)                            // pad_enable
              | (pad_int << 22) | (pad_amt << 25));
  g1[1] = (int)(tile_k << 16);                        // tensor_dim0 lo16
  g1[2] = (int)(rows << 16);                          // tensor_dim0 hi | tensor_dim1 lo
  g1[3] = (int)(tile_k << 16);                        // tensor_dim1 hi | tile_dim0
  g1[4] = (int)rows;                                  // tile_dim1 (tile_dim2=0 => 2D)
  g1[5] = (int)stride_elems;                          // tensor_dim0_stride lo32
  g1[6] = 0;
  g1[7] = 0;
  i32x4 gz = {0, 0, 0, 0};
#if __has_include(<hip/amd_detail/amd_gfx1250_TDM.h>)
  i32x8 gz8 = {0, 0, 0, 0, 0, 0, 0, 0};
  __builtin_amdgcn_tensor_load_to_lds(g0, g1, gz, gz, gz8, 0);
#else
  __builtin_amdgcn_tensor_load_to_lds(g0, g1, gz, gz, 0);
#endif
}
#endif

// ---------------------------------------------------------------------------
// One-shot weight transpose+convert: W fp32 [k][n] -> Wt bf16 [n][k]
// ---------------------------------------------------------------------------
__global__ __launch_bounds__(256) void wt_conv_kernel(const float* __restrict__ W,
                                                      unsigned short* __restrict__ Wt) {
  const int idx = blockIdx.x * 256 + threadIdx.x;     // over 1024*1024
  const int k = idx >> 10, n = idx & 1023;
  Wt[(size_t)n * D_MODEL + k] = f2bf(W[idx]);
}

// ---------------------------------------------------------------------------
// Projection GEMM: Y = (X @ W + b) * scale ; X fp32 (8192x1024), Wt bf16
// [n][k]. Output bf16 head-split (B,H,L,DK), or transposed (B,H,DK,L) when
// vtrans!=0 (used for V so attention's PV B-matrix needs no LDS transpose).
// 128x64 tile per WG, 4 waves x (32 rows x 64 cols), double-buffered LDS.
// ---------------------------------------------------------------------------
__global__ __launch_bounds__(128) void proj_qkv_kernel(
    const float* __restrict__ X, const unsigned short* __restrict__ Wt,
    const float* __restrict__ bias, unsigned short* __restrict__ Y,
    float scale, int vtrans) {
  const int nb = blockIdx.x;   // 0..15  (== head index, N-tile = DK)
  const int mb = blockIdx.y;   // 0..63
  const int tid = threadIdx.x;
  const int wave = tid >> 5;
  const int lane = tid & 31;
  const int ln = lane & 15;
  const int h  = lane >> 4;

  __shared__ __align__(16) unsigned short sA[2][128 * 40];
  __shared__ __align__(16) unsigned short sB[2][64 * 40];

  v8f acc[2][4] = {};
  const float* Xtile = X + (size_t)mb * 128 * D_MODEL;
#if USE_TDM
  const int wv = __builtin_amdgcn_readfirstlane(wave);
#endif

  auto stageA = [&](int b, int kb) {    // fp32 -> bf16, one row per thread
    const float* src = Xtile + (size_t)tid * D_MODEL + kb * 32;
    unsigned short* dst = &sA[b][tid * 40];
#pragma unroll
    for (int j = 0; j < 8; ++j) {
      float4 f = *(const float4*)(src + j * 4);
      *(unsigned int*)(dst + j * 4)     = f2bf2(f.x, f.y);
      *(unsigned int*)(dst + j * 4 + 2) = f2bf2(f.z, f.w);
    }
    if (kb + 1 < D_MODEL / 32) __builtin_prefetch(src + 32, 0, 3);
  };
  auto stageB = [&](int b, int kb) {    // bf16 Wt tile copy
#if USE_TDM
    tdm_load_2d(Wt + (size_t)(nb * 64 + wv * 16) * D_MODEL + kb * 32,
                (unsigned)(uintptr_t)&sB[b][wv * 16 * 40], 32, 16, D_MODEL, 3, 3);
#else
    const int n = tid >> 1, half = tid & 1;
    const unsigned short* src = Wt + (size_t)(nb * 64 + n) * D_MODEL + kb * 32 + half * 16;
    *(uint4*)&sB[b][n * 40 + half * 16] = *(const uint4*)src;
    *(uint4*)&sB[b][n * 40 + half * 16 + 8] = *(const uint4*)(src + 8);
#endif
  };

  stageA(0, 0);
  stageB(0, 0);
  for (int kb = 0; kb < D_MODEL / 32; ++kb) {
    const int cur = kb & 1;
#if USE_TDM
    __builtin_amdgcn_s_wait_tensorcnt(0);
#endif
    __syncthreads();
    if (kb + 1 < D_MODEL / 32) { stageA(cur ^ 1, kb + 1); stageB(cur ^ 1, kb + 1); }

    v16bf fa[2];
#pragma unroll
    for (int u = 0; u < 2; ++u) {
      const int m = wave * 32 + u * 16 + ln;
      fa[u] = ld_frag(&sA[cur][m * 40 + h * 8], &sA[cur][m * 40 + 16 + h * 8]);
    }
#pragma unroll
    for (int t = 0; t < 4; ++t) {
      const int n = t * 16 + ln;
      v16bf fb = ld_frag(&sB[cur][n * 40 + h * 8], &sB[cur][n * 40 + 16 + h * 8]);
      acc[0][t] = WMMA_BF16(fa[0], fb, acc[0][t]);
      acc[1][t] = WMMA_BF16(fa[1], fb, acc[1][t]);
    }
  }

  // ---- epilogue: bias + scale, write head-split bf16
#pragma unroll
  for (int u = 0; u < 2; ++u) {
#pragma unroll
    for (int t = 0; t < 4; ++t) {
      const int d = t * 16 + ln;
      const float bv = bias[nb * 64 + d];
#pragma unroll
      for (int r = 0; r < 8; ++r) {
        const int row = mb * 128 + wave * 32 + u * 16 + h * 8 + r;  // 0..8191
        const int bi = row >> 11;
        const int l  = row & (SEQLEN - 1);
        const float v = (acc[u][t][r] + bv) * scale;
        const size_t off = vtrans
            ? ((size_t)(bi * NHEAD + nb) * DK + d) * SEQLEN + l
            : ((size_t)(bi * NHEAD + nb) * SEQLEN + l) * DK + d;
        Y[off] = f2bf(v);
      }
    }
  }
}

// ---------------------------------------------------------------------------
// Flash attention (causal): one workgroup = one (batch*head, 64-query tile).
// Qh/Kh bf16 (B*H, L, 64); Vt bf16 (B*H, 64, L). Ctx bf16 (B, L, D_MODEL).
// Double-buffered K/V tiles; TDM DMA with hardware row padding when present.
// ---------------------------------------------------------------------------
__global__ __launch_bounds__(128) void flash_attn_kernel(
    const unsigned short* __restrict__ Qh, const unsigned short* __restrict__ Kh,
    const unsigned short* __restrict__ Vt, unsigned short* __restrict__ Ctx) {
  const int bh = blockIdx.x;   // 0..63
  const int qt = blockIdx.y;   // 0..31
  const int tid = threadIdx.x;
  const int wave = tid >> 5;
  const int lane = tid & 31;
  const int ln = lane & 15;
  const int h  = lane >> 4;
  const int qb = qt * 64;

  __shared__ __align__(16) unsigned short sQ [64 * 72];     // [q][d]
  __shared__ __align__(16) unsigned short sK [2][64 * 72];  // [key][d]
  __shared__ __align__(16) unsigned short sVt[2][64 * 72];  // [d][key]
  __shared__ __align__(16) unsigned short sP [64 * 72];     // [q][key] per-wave rows

  const unsigned short* Qb  = Qh + (size_t)bh * SEQLEN * DK;
  const unsigned short* Kb  = Kh + (size_t)bh * SEQLEN * DK;
  const unsigned short* Vtb = Vt + (size_t)bh * DK * SEQLEN;

#if USE_TDM
  const int wv = __builtin_amdgcn_readfirstlane(wave);
#endif

  auto stageK = [&](int b, int kb2) {
    const int kbase2 = kb2 * 64;
#if USE_TDM
    tdm_load_2d(Kb + (size_t)(kbase2 + wv * 16) * DK,
                (unsigned)(uintptr_t)&sK[b][wv * 16 * 72], 64, 16, 64, 4, 3);
#else
    const int r = tid >> 1, half = tid & 1;
    const unsigned short* src = Kb + (size_t)(kbase2 + r) * DK + half * 32;
#pragma unroll
    for (int j = 0; j < 4; ++j)
      *(uint4*)&sK[b][r * 72 + half * 32 + j * 8] = *(const uint4*)(src + j * 8);
#endif
  };
  auto stageV = [&](int b, int kb2) {   // Vt rows are d; keys contiguous
    const int kbase2 = kb2 * 64;
#if USE_TDM
    tdm_load_2d(Vtb + (size_t)(wv * 16) * SEQLEN + kbase2,
                (unsigned)(uintptr_t)&sVt[b][wv * 16 * 72], 64, 16, SEQLEN, 4, 3);
#else
    const int d = tid >> 1, half = tid & 1;
    const unsigned short* src = Vtb + (size_t)d * SEQLEN + kbase2 + half * 32;
#pragma unroll
    for (int j = 0; j < 4; ++j)
      *(uint4*)&sVt[b][d * 72 + half * 32 + j * 8] = *(const uint4*)(src + j * 8);
#endif
  };

  // ---- stage Q tile (64x64 bf16)
#if USE_TDM
  tdm_load_2d(Qb + (size_t)(qb + wv * 16) * DK,
              (unsigned)(uintptr_t)&sQ[wv * 16 * 72], 64, 16, 64, 4, 3);
  __builtin_amdgcn_s_wait_tensorcnt(0);
#else
  {
    const int r = tid >> 1, half = tid & 1;
    const unsigned short* src = Qb + (size_t)(qb + r) * DK + half * 32;
#pragma unroll
    for (int j = 0; j < 4; ++j)
      *(uint4*)&sQ[r * 72 + half * 32 + j * 8] = *(const uint4*)(src + j * 8);
  }
#endif
  __syncthreads();

  // per-wave Q A-fragments (two K=32 chunks over dk=64), kept in VGPRs
  const int mq = wave * 16 + ln;
  v16bf qa[2];
#pragma unroll
  for (int c = 0; c < 2; ++c)
    qa[c] = ld_frag(&sQ[mq * 72 + c * 32 + h * 8], &sQ[mq * 72 + c * 32 + 16 + h * 8]);

  v8f oacc[4] = {};
  float mrun[8], lrun[8];
#pragma unroll
  for (int r = 0; r < 8; ++r) { mrun[r] = -3.0e38f; lrun[r] = 0.0f; }

  const int nkb = qt + 1;  // causal: only key blocks <= query block
  stageK(0, 0);
  stageV(0, 0);
  for (int kb = 0; kb < nkb; ++kb) {
    const int kbase = kb * 64;
    const int cur = kb & 1;
#if USE_TDM
    __builtin_amdgcn_s_wait_tensorcnt(0);
#endif
    __syncthreads();
    if (kb + 1 < nkb) { stageK(cur ^ 1, kb + 1); stageV(cur ^ 1, kb + 1); }

    // ---- scores S = Q @ K^T  (Q pre-scaled by 1/sqrt(dk))
    v8f s[4];
#pragma unroll
    for (int t = 0; t < 4; ++t) {
      v8f a = {};
      const int n = t * 16 + ln;
#pragma unroll
      for (int c = 0; c < 2; ++c) {
        v16bf fb = ld_frag(&sK[cur][n * 72 + c * 32 + h * 8],
                           &sK[cur][n * 72 + c * 32 + 16 + h * 8]);
        a = WMMA_BF16(qa[c], fb, a);
      }
      s[t] = a;
    }

    // ---- causal mask (only diagonal block needs it)
    if (kb == qt) {
#pragma unroll
      for (int t = 0; t < 4; ++t) {
        const int kg = kbase + t * 16 + ln;
#pragma unroll
        for (int r = 0; r < 8; ++r) {
          const int qg = qb + wave * 16 + h * 8 + r;
          if (kg > qg) s[t][r] = -3.0e38f;
        }
      }
    }

    // ---- online softmax (row = VGPR index r within 16-lane half)
#pragma unroll
    for (int r = 0; r < 8; ++r) {
      float v = fmaxf(fmaxf(s[0][r], s[1][r]), fmaxf(s[2][r], s[3][r]));
#pragma unroll
      for (int off = 1; off < 16; off <<= 1)
        v = fmaxf(v, __shfl_xor(v, off, 32));
      const float mnew  = fmaxf(mrun[r], v);
      const float alpha = __expf(mrun[r] - mnew);
      float rsum = 0.0f;
#pragma unroll
      for (int t = 0; t < 4; ++t) {
        const float p = __expf(s[t][r] - mnew);
        s[t][r] = p;
        rsum += p;
      }
#pragma unroll
      for (int off = 1; off < 16; off <<= 1)
        rsum += __shfl_xor(rsum, off, 32);
      lrun[r] = lrun[r] * alpha + rsum;
      mrun[r] = mnew;
#pragma unroll
      for (int t = 0; t < 4; ++t) oacc[t][r] *= alpha;
    }

    // ---- write P (C layout) to per-wave LDS rows, re-read as A fragments
#pragma unroll
    for (int t = 0; t < 4; ++t)
#pragma unroll
      for (int r = 0; r < 8; ++r)
        sP[(wave * 16 + h * 8 + r) * 72 + t * 16 + ln] = f2bf(s[t][r]);

    // ---- O += P @ V
#pragma unroll
    for (int c = 0; c < 2; ++c) {
      v16bf pa = ld_frag(&sP[(wave * 16 + ln) * 72 + c * 32 + h * 8],
                         &sP[(wave * 16 + ln) * 72 + c * 32 + 16 + h * 8]);
#pragma unroll
      for (int t = 0; t < 4; ++t) {
        const int n = t * 16 + ln;
        v16bf vb = ld_frag(&sVt[cur][n * 72 + c * 32 + h * 8],
                           &sVt[cur][n * 72 + c * 32 + 16 + h * 8]);
        oacc[t] = WMMA_BF16(pa, vb, oacc[t]);
      }
    }
  }

  // ---- epilogue: normalize and write ctx bf16 (B, L, D_MODEL)
  const int bi = bh >> 4, head = bh & 15;
#pragma unroll
  for (int t = 0; t < 4; ++t) {
    const int d = t * 16 + ln;
#pragma unroll
    for (int r = 0; r < 8; ++r) {
      const int qg = qb + wave * 16 + h * 8 + r;
      const float inv = 1.0f / lrun[r];
      const size_t off = ((size_t)(bi * SEQLEN + qg)) * D_MODEL + head * DK + d;
      Ctx[off] = f2bf(oacc[t][r] * inv);
    }
  }
}

// ---------------------------------------------------------------------------
// Output projection: out = ctx(bf16) @ Wto(bf16 [n][k]) + bo, out fp32
// 128x64 tile per WG, double-buffered; A and B staged via TDM when present.
// ---------------------------------------------------------------------------
__global__ __launch_bounds__(128) void out_proj_kernel(
    const unsigned short* __restrict__ Xb, const unsigned short* __restrict__ Wt,
    const float* __restrict__ bias, float* __restrict__ Y) {
  const int nb = blockIdx.x;
  const int mb = blockIdx.y;
  const int tid = threadIdx.x;
  const int wave = tid >> 5;
  const int lane = tid & 31;
  const int ln = lane & 15;
  const int h  = lane >> 4;

  __shared__ __align__(16) unsigned short sA[2][128 * 40];
  __shared__ __align__(16) unsigned short sB[2][64 * 40];

  v8f acc[2][4] = {};
#if USE_TDM
  const int wv = __builtin_amdgcn_readfirstlane(wave);
#endif

  auto stageA = [&](int b, int kb) {
#if USE_TDM
    tdm_load_2d(Xb + (size_t)(mb * 128 + wv * 32) * D_MODEL + kb * 32,
                (unsigned)(uintptr_t)&sA[b][wv * 32 * 40], 32, 32, D_MODEL, 3, 3);
#else
    const unsigned short* src = Xb + (size_t)(mb * 128 + tid) * D_MODEL + kb * 32;
#pragma unroll
    for (int j = 0; j < 4; ++j)
      *(uint4*)&sA[b][tid * 40 + j * 8] = *(const uint4*)(src + j * 8);
#endif
  };
  auto stageB = [&](int b, int kb) {
#if USE_TDM
    tdm_load_2d(Wt + (size_t)(nb * 64 + wv * 16) * D_MODEL + kb * 32,
                (unsigned)(uintptr_t)&sB[b][wv * 16 * 40], 32, 16, D_MODEL, 3, 3);
#else
    const int n = tid >> 1, half = tid & 1;
    const unsigned short* src = Wt + (size_t)(nb * 64 + n) * D_MODEL + kb * 32 + half * 16;
    *(uint4*)&sB[b][n * 40 + half * 16] = *(const uint4*)src;
    *(uint4*)&sB[b][n * 40 + half * 16 + 8] = *(const uint4*)(src + 8);
#endif
  };

  stageA(0, 0);
  stageB(0, 0);
  for (int kb = 0; kb < D_MODEL / 32; ++kb) {
    const int cur = kb & 1;
#if USE_TDM
    __builtin_amdgcn_s_wait_tensorcnt(0);
#endif
    __syncthreads();
    if (kb + 1 < D_MODEL / 32) { stageA(cur ^ 1, kb + 1); stageB(cur ^ 1, kb + 1); }

    v16bf fa[2];
#pragma unroll
    for (int u = 0; u < 2; ++u) {
      const int m = wave * 32 + u * 16 + ln;
      fa[u] = ld_frag(&sA[cur][m * 40 + h * 8], &sA[cur][m * 40 + 16 + h * 8]);
    }
#pragma unroll
    for (int t = 0; t < 4; ++t) {
      const int n = t * 16 + ln;
      v16bf fb = ld_frag(&sB[cur][n * 40 + h * 8], &sB[cur][n * 40 + 16 + h * 8]);
      acc[0][t] = WMMA_BF16(fa[0], fb, acc[0][t]);
      acc[1][t] = WMMA_BF16(fa[1], fb, acc[1][t]);
    }
  }

#pragma unroll
  for (int u = 0; u < 2; ++u) {
#pragma unroll
    for (int t = 0; t < 4; ++t) {
      const int n = nb * 64 + t * 16 + ln;
      const float bv = bias[n];
#pragma unroll
      for (int r = 0; r < 8; ++r) {
        const int row = mb * 128 + wave * 32 + u * 16 + h * 8 + r;
        Y[(size_t)row * D_MODEL + n] = acc[u][t][r] + bv;
      }
    }
  }
}

// ---------------------------------------------------------------------------
extern "C" void kernel_launch(void* const* d_in, const int* in_sizes, int n_in,
                              void* d_out, int out_size, void* d_ws, size_t ws_size,
                              hipStream_t stream) {
  (void)in_sizes; (void)n_in; (void)out_size; (void)ws_size;
  const float* q  = (const float*)d_in[0];
  const float* k  = (const float*)d_in[1];
  const float* v  = (const float*)d_in[2];
  // d_in[3]: attn_mask (bool, causal) -- realized analytically in-kernel
  const float* Wq = (const float*)d_in[4];
  const float* bq = (const float*)d_in[5];
  const float* Wk = (const float*)d_in[6];
  const float* bk = (const float*)d_in[7];
  const float* Wv = (const float*)d_in[8];
  const float* bv = (const float*)d_in[9];
  const float* Wo = (const float*)d_in[10];
  const float* bo = (const float*)d_in[11];
  float* out = (float*)d_out;

  const size_t headElems = (size_t)BATCH * SEQLEN * D_MODEL;  // bf16 elems each
  const size_t wElems    = (size_t)D_MODEL * D_MODEL;
  unsigned short* Qh  = (unsigned short*)d_ws;
  unsigned short* Kh  = Qh + headElems;
  unsigned short* Vt  = Kh + headElems;        // V stored (B,H,DK,L)
  unsigned short* Ctx = Vt + headElems;
  unsigned short* Wtq = Ctx + headElems;       // transposed bf16 weights
  unsigned short* Wtk = Wtq + wElems;
  unsigned short* Wtv = Wtk + wElems;
  unsigned short* Wto = Wtv + wElems;

  const dim3 blk(128);
  const dim3 gProj(D_MODEL / 64, (BATCH * SEQLEN) / 128);   // (16, 64)
  const dim3 gConv(wElems / 256);

  wt_conv_kernel<<<gConv, 256, 0, stream>>>(Wq, Wtq);
  wt_conv_kernel<<<gConv, 256, 0, stream>>>(Wk, Wtk);
  wt_conv_kernel<<<gConv, 256, 0, stream>>>(Wv, Wtv);
  wt_conv_kernel<<<gConv, 256, 0, stream>>>(Wo, Wto);

  proj_qkv_kernel<<<gProj, blk, 0, stream>>>(q, Wtq, bq, Qh, 0.125f, 0);  // 1/sqrt(64)
  proj_qkv_kernel<<<gProj, blk, 0, stream>>>(k, Wtk, bk, Kh, 1.0f, 0);
  proj_qkv_kernel<<<gProj, blk, 0, stream>>>(v, Wtv, bv, Vt, 1.0f, 1);    // transposed

  flash_attn_kernel<<<dim3(BATCH * NHEAD, SEQLEN / 64), blk, 0, stream>>>(Qh, Kh, Vt, Ctx);

  out_proj_kernel<<<gProj, blk, 0, stream>>>(Ctx, Wto, bo, out);
}
